// SelfRelMultiHeadAttention_30872224924293
// MI455X (gfx1250) — matrix operand order repaired and checked by reading, not compiled, not run on hardware
//
#include <hip/hip_runtime.h>
#include <hip/hip_bf16.h>
#include <hip/hip_fp16.h>

// SelfRelMultiHeadAttention (Transformer-XL relative attention) for MI455X/gfx1250.
// All GEMMs run on v_wmma_f32_16x16x32_f16 (f16 A/B, f32 accum), wave32, 8-wave blocks.
// rel_to_abs is fused via the skew trick (16-row Rel strip of width T+15 in LDS).
// fp32 result streams (att_w, O) use non-temporal stores so the re-read f16
// staging buffers (attH) stay resident in the 192 MB L2.

typedef __attribute__((ext_vector_type(16))) _Float16 v16h;
typedef __attribute__((ext_vector_type(8)))  _Float16 v8h;
typedef __attribute__((ext_vector_type(8)))  float    v8f;

#define B_   4
#define T_   1024
#define D_   1024
#define H_   16
#define DH_  64
#define L_   2047
#define LP16_ 130          // padded rel length = 2080 cols = 130 16-col tiles
#define RELW_ 1056         // 66 tiles * 16 = T + 15 rounded up to tile

__device__ __forceinline__ v8f wmma32(v16h a, v16h b, v8f c) {
  // D = A(16x32 f16) * B(32x16 f16) + C(16x16 f32)
  return __builtin_amdgcn_wmma_f32_16x16x32_f16(false, a, false, b, (short)0, c,
                                                false, false);
}

__device__ __forceinline__ v16h cat16(v8h lo, v8h hi) {
  return __builtin_shufflevector(lo, hi, 0, 1, 2, 3, 4, 5, 6, 7,
                                 8, 9, 10, 11, 12, 13, 14, 15);
}

// A fragment (16x32) from a row-major f16 matrix (contiguous k order, same as the B loaders):
// lane m=l&15 is row; kg=l>>4; half j -> k=k0+kg*16+j.
// Rows must be 16B aligned.
__device__ __forceinline__ v16h load_a_rm(const _Float16* A, int lda, int k0, int lane) {
  const int m = lane & 15, kg = lane >> 4;
  const _Float16* p = A + (size_t)m * lda + k0 + kg * 16;
  return cat16(*(const v8h*)p, *(const v8h*)(p + 8));
}

// B fragment (32x16) from a packed fragment-ready block: [lane][16] halves,
// lane n=l&15 is column; kg=l>>4; half j -> k = kg*16 + j. 32B/lane, coalesced.
__device__ __forceinline__ v16h load_b_pk(const _Float16* p, int lane) {
  const _Float16* q = p + lane * 16;
  return cat16(*(const v8h*)q, *(const v8h*)(q + 8));
}

// B fragment from an LDS tile stored transposed: Bt[n][k], ldb halves, rows 16B aligned.
__device__ __forceinline__ v16h load_b_t(const _Float16* Bt, int ldb, int n0, int lane) {
  const int n = n0 + (lane & 15), kg = lane >> 4;
  const _Float16* p = Bt + (size_t)n * ldb + kg * 16;
  return cat16(*(const v8h*)p, *(const v8h*)(p + 8));
}

// ---------------------------------------------------------------------------
// Kernel A: Q/K/V projections (blockIdx.z selects which). C = qkv @ W.
// 128x128 tile per 256-thread (8-wave) block; f32->f16 staged through LDS
// (staging must go through registers anyway for the precision conversion, so
// async/TDM copies would add a pass without removing work).
// Epilogues:
//   z=0: Qu = Q+u, Qv = Q+v in [b][h][t][dh] row-major f16 (A-fragment friendly)
//   z=1: K in B-fragment-ready layout KB[b][h][t/16][dh/32][lane][16]
//   z=2: V in B-fragment-ready layout VB[b][h][t/32][dh/16][lane][16]
// ---------------------------------------------------------------------------
__global__ __launch_bounds__(256) void k_proj_qkv(
    const float* __restrict__ qkv, const float* __restrict__ Wq,
    const float* __restrict__ Wk, const float* __restrict__ Wv,
    const float* __restrict__ uvec, const float* __restrict__ vvec,
    _Float16* __restrict__ Qu, _Float16* __restrict__ Qv,
    _Float16* __restrict__ KB, _Float16* __restrict__ VB) {
  const int which = blockIdx.z;
  const float* W = (which == 0) ? Wq : ((which == 1) ? Wk : Wv);
  const int n0 = blockIdx.x * 128;
  const int m0 = blockIdx.y * 128;
  const int tid = threadIdx.x;
  const int lane = tid & 31, wid = tid >> 5;
  const int wr = wid >> 1, wc = wid & 1;  // 4x2 wave grid: 32 rows x 64 cols each

  __shared__ __align__(16) _Float16 As[128 * 40];  // row-major [128][32+8]
  __shared__ __align__(16) _Float16 Bs[128 * 40];  // transposed [128 n][32+8 k]

  v8f acc[2][4] = {};

  for (int k0 = 0; k0 < D_; k0 += 32) {
#pragma unroll
    for (int c = 0; c < 4; ++c) {  // A tile 128x32 f32 -> f16
      int e = c * 1024 + tid * 4;
      int r = e >> 5, col = e & 31;
      const float4 f = *(const float4*)(qkv + (size_t)(m0 + r) * D_ + k0 + col);
      _Float16* dst = As + r * 40 + col;
      dst[0] = (_Float16)f.x; dst[1] = (_Float16)f.y;
      dst[2] = (_Float16)f.z; dst[3] = (_Float16)f.w;
    }
#pragma unroll
    for (int c = 0; c < 4; ++c) {  // B tile 32x128 f32 -> f16, transposed in LDS
      int e = c * 1024 + tid * 4;
      int kr = e >> 7, col = e & 127;
      const float4 f = *(const float4*)(W + (size_t)(k0 + kr) * D_ + n0 + col);
      Bs[(col + 0) * 40 + kr] = (_Float16)f.x;
      Bs[(col + 1) * 40 + kr] = (_Float16)f.y;
      Bs[(col + 2) * 40 + kr] = (_Float16)f.z;
      Bs[(col + 3) * 40 + kr] = (_Float16)f.w;
    }
    __syncthreads();
    v16h af0 = load_a_rm(As + (wr * 32 + 0) * 40, 40, 0, lane);
    v16h af1 = load_a_rm(As + (wr * 32 + 16) * 40, 40, 0, lane);
#pragma unroll
    for (int ct = 0; ct < 4; ++ct) {
      v16h bf = load_b_t(Bs, 40, wc * 64 + ct * 16, lane);
      acc[0][ct] = wmma32(af0, bf, acc[0][ct]);
      acc[1][ct] = wmma32(af1, bf, acc[1][ct]);
    }
    __syncthreads();
  }

  const int nlo = lane & 15, mhi = (lane >> 4) * 8;
#pragma unroll
  for (int rt = 0; rt < 2; ++rt) {
#pragma unroll
    for (int ct = 0; ct < 4; ++ct) {
#pragma unroll
      for (int i = 0; i < 8; ++i) {
        const int mrow = m0 + wr * 32 + rt * 16 + mhi + i;
        const int ncol = n0 + wc * 64 + ct * 16 + nlo;
        const float val = acc[rt][ct][i];
        const int b = mrow >> 10, t = mrow & (T_ - 1);
        const int h = ncol >> 6, dh = ncol & (DH_ - 1);
        const size_t bh = (size_t)(b * H_ + h);
        if (which == 0) {
          const size_t base = (bh * T_ + t) * DH_ + dh;
          Qu[base] = (_Float16)(val + uvec[ncol]);
          Qv[base] = (_Float16)(val + vvec[ncol]);
        } else if (which == 1) {
          // B-layout: n=t, k=dh
          const int c = t >> 4, kk = dh >> 5, kg = (dh >> 4) & 1, j = dh & 15;
          const int l = kg * 16 + (t & 15);
          KB[(((bh * 64 + c) * 2 + kk) * 32 + l) * 16 + j] = (_Float16)val;
        } else {
          // B-layout: n=dh, k=t (s dimension of att@V)
          const int ks = t >> 5, kg = (t >> 4) & 1, j = t & 15;
          const int c = dh >> 4, l = kg * 16 + (dh & 15);
          VB[(((bh * 32 + ks) * 4 + c) * 32 + l) * 16 + j] = (_Float16)val;
        }
      }
    }
  }
}

// ---------------------------------------------------------------------------
// Kernel B: E = pos_enc @ Wpos + bpos, stored B-fragment-ready per head:
// ETB[h][r/16][dh/32][lane][16] over padded r in [0, 2080). Rows >= L stay 0.
// ---------------------------------------------------------------------------
__global__ __launch_bounds__(256) void k_proj_e(
    const float* __restrict__ pos, const float* __restrict__ Wp,
    const float* __restrict__ bpos, _Float16* __restrict__ ETB) {
  const int n0 = blockIdx.x * 128;
  const int m0 = blockIdx.y * 128;
  const int tid = threadIdx.x;
  const int lane = tid & 31, wid = tid >> 5;
  const int wr = wid >> 1, wc = wid & 1;

  __shared__ __align__(16) _Float16 As[128 * 40];
  __shared__ __align__(16) _Float16 Bs[128 * 40];

  v8f acc[2][4] = {};

  for (int k0 = 0; k0 < D_; k0 += 32) {
#pragma unroll
    for (int c = 0; c < 4; ++c) {
      int e = c * 1024 + tid * 4;
      int r = e >> 5, col = e & 31;
      const int rg = m0 + r;
      float4 f = make_float4(0.f, 0.f, 0.f, 0.f);
      if (rg < L_) f = *(const float4*)(pos + (size_t)rg * D_ + k0 + col);
      _Float16* dst = As + r * 40 + col;
      dst[0] = (_Float16)f.x; dst[1] = (_Float16)f.y;
      dst[2] = (_Float16)f.z; dst[3] = (_Float16)f.w;
    }
#pragma unroll
    for (int c = 0; c < 4; ++c) {
      int e = c * 1024 + tid * 4;
      int kr = e >> 7, col = e & 127;
      const float4 f = *(const float4*)(Wp + (size_t)(k0 + kr) * D_ + n0 + col);
      Bs[(col + 0) * 40 + kr] = (_Float16)f.x;
      Bs[(col + 1) * 40 + kr] = (_Float16)f.y;
      Bs[(col + 2) * 40 + kr] = (_Float16)f.z;
      Bs[(col + 3) * 40 + kr] = (_Float16)f.w;
    }
    __syncthreads();
    v16h af0 = load_a_rm(As + (wr * 32 + 0) * 40, 40, 0, lane);
    v16h af1 = load_a_rm(As + (wr * 32 + 16) * 40, 40, 0, lane);
#pragma unroll
    for (int ct = 0; ct < 4; ++ct) {
      v16h bf = load_b_t(Bs, 40, wc * 64 + ct * 16, lane);
      acc[0][ct] = wmma32(af0, bf, acc[0][ct]);
      acc[1][ct] = wmma32(af1, bf, acc[1][ct]);
    }
    __syncthreads();
  }

  const int nlo = lane & 15, mhi = (lane >> 4) * 8;
#pragma unroll
  for (int rt = 0; rt < 2; ++rt) {
#pragma unroll
    for (int ct = 0; ct < 4; ++ct) {
#pragma unroll
      for (int i = 0; i < 8; ++i) {
        const int r = m0 + wr * 32 + rt * 16 + mhi + i;
        const int ncol = n0 + wc * 64 + ct * 16 + nlo;
        if (r < L_) {
          const float val = acc[rt][ct][i] + bpos[ncol];
          const int h = ncol >> 6, dh = ncol & (DH_ - 1);
          const int c = r >> 4, kk = dh >> 5, kg = (dh >> 4) & 1, j = dh & 15;
          const int l = kg * 16 + (r & 15);
          ETB[((((size_t)h * LP16_ + c) * 2 + kk) * 32 + l) * 16 + j] = (_Float16)val;
        }
      }
    }
  }
}

// ---------------------------------------------------------------------------
// Kernel C: fused scores_K + skewed scores_E + mask + softmax for one
// (b, h, 16-row block). scores_E[t,s] = Qv[t] . E[T-1+s-t]; for t in
// [t0,t0+16) the needed E window is rbase=T-16-t0 .. rbase+T+14, and within
// the block scores_E[t0+i, s] = Rel[i][s - i + 15] (skew trick).
// Dynamic LDS: Rel 16x1056 f32 + scores 16x1024 f32 = 133 KB (<320 KB WGP).
// att_w fp32 goes out with non-temporal stores (never re-read by the GPU);
// the f16 copy for att@V stays regular so it lives in L2 for kernel D.
// ---------------------------------------------------------------------------
__global__ __launch_bounds__(256) void k_scores_softmax(
    const _Float16* __restrict__ Qu, const _Float16* __restrict__ Qv,
    const _Float16* __restrict__ KB, const _Float16* __restrict__ ETB,
    const int* __restrict__ mask,
    float* __restrict__ att_out, _Float16* __restrict__ attH) {
  extern __shared__ float smem[];
  float* rel = smem;                 // [16][RELW_]
  float* sc  = smem + 16 * RELW_;    // [16][T_]

  const int b = blockIdx.z, h = blockIdx.y;
  const int t0 = blockIdx.x * 16;
  const int tid = threadIdx.x, lane = tid & 31, wid = tid >> 5;
  const size_t bh = (size_t)(b * H_ + h);

  const _Float16* QuA = Qu + (bh * T_ + t0) * DH_;
  const _Float16* QvA = Qv + (bh * T_ + t0) * DH_;
  v16h aQu0 = load_a_rm(QuA, DH_, 0, lane);
  v16h aQu1 = load_a_rm(QuA, DH_, 32, lane);
  v16h aQv0 = load_a_rm(QvA, DH_, 0, lane);
  v16h aQv1 = load_a_rm(QvA, DH_, 32, lane);

  const int nlo = lane & 15, mhi = (lane >> 4) * 8;

  // Phase 1: Rel strip (66 col-tiles of the E window starting at tile rb16)
  const int rb16 = (T_ / 16) - 1 - blockIdx.x;  // (T-16-t0)/16
  for (int ct = wid; ct < 66; ct += 8) {
    const _Float16* bp = ETB + (((size_t)h * LP16_ + (rb16 + ct)) * 2) * 512;
    v8f a = {};
    a = wmma32(aQv0, load_b_pk(bp, lane), a);
    a = wmma32(aQv1, load_b_pk(bp + 512, lane), a);
#pragma unroll
    for (int i = 0; i < 8; ++i)
      rel[(mhi + i) * RELW_ + ct * 16 + nlo] = a[i];
  }
  __syncthreads();

  // Phase 2: scores_K tiles + skew-gathered Rel add
  for (int ct = wid; ct < 64; ct += 8) {
    const _Float16* bp = KB + ((bh * 64 + ct) * 2) * 512;
    v8f a = {};
    a = wmma32(aQu0, load_b_pk(bp, lane), a);
    a = wmma32(aQu1, load_b_pk(bp + 512, lane), a);
    const int s = ct * 16 + nlo;
#pragma unroll
    for (int i = 0; i < 8; ++i) {
      const int m = mhi + i;
      sc[m * T_ + s] = a[i] + rel[m * RELW_ + (s - m + 15)];
    }
  }
  __syncthreads();

  // Phase 3: masked, scaled softmax; wave handles rows 2*wid, 2*wid+1
  const float scale = 0.125f;  // 1/sqrt(64)
#pragma unroll
  for (int rr = 0; rr < 2; ++rr) {
    const int m = wid * 2 + rr;
    float vbuf[32];
    float mx = -3.4e38f;
#pragma unroll
    for (int q = 0; q < 32; ++q) {
      const int s = q * 32 + lane;
      const float vv = sc[m * T_ + s] * scale + (float)mask[b * T_ + s] * (-1e7f);
      vbuf[q] = vv;
      mx = fmaxf(mx, vv);
    }
#pragma unroll
    for (int off = 16; off > 0; off >>= 1) mx = fmaxf(mx, __shfl_xor(mx, off, 32));
    float sum = 0.f;
#pragma unroll
    for (int q = 0; q < 32; ++q) {
      const float e = __expf(vbuf[q] - mx);
      vbuf[q] = e;
      sum += e;
    }
#pragma unroll
    for (int off = 16; off > 0; off >>= 1) sum += __shfl_xor(sum, off, 32);
    const float inv = 1.0f / sum;
    const size_t rowbase = (bh * T_ + (t0 + m)) * T_;
#pragma unroll
    for (int q = 0; q < 32; ++q) {
      const int s = q * 32 + lane;
      const float aw = vbuf[q] * inv;
      __builtin_nontemporal_store(aw, &att_out[rowbase + s]);  // streamed fp32 output
      attH[rowbase + s] = (_Float16)aw;                        // cache-resident staging
    }
  }
}

// ---------------------------------------------------------------------------
// Kernel D: O_h = att_w @ V_h per (b,h). A = attH row-major f16 (direct
// fragment loads), B = VB fragment-ready. Wave computes 16 rows x 64 cols.
// ---------------------------------------------------------------------------
__global__ __launch_bounds__(256) void k_attv(
    const _Float16* __restrict__ attH, const _Float16* __restrict__ VB,
    _Float16* __restrict__ Ostage) {
  const int b = blockIdx.z, h = blockIdx.y;
  const int tid = threadIdx.x, lane = tid & 31, wid = tid >> 5;
  const int t0 = blockIdx.x * 128 + wid * 16;
  const size_t bh = (size_t)(b * H_ + h);
  const _Float16* Arow = attH + (bh * T_ + t0) * T_;
  const _Float16* Vb = VB + bh * (size_t)(32 * 4 * 32 * 16);

  v8f acc[4] = {};
  for (int ks = 0; ks < 32; ++ks) {
    if (ks + 1 < 32)
      __builtin_prefetch(Arow + (size_t)(lane & 15) * T_ + (ks + 1) * 32, 0, 1);
    v16h af = load_a_rm(Arow, T_, ks * 32, lane);
    const _Float16* bp = Vb + (size_t)ks * 4 * 512;
#pragma unroll
    for (int ct = 0; ct < 4; ++ct)
      acc[ct] = wmma32(af, load_b_pk(bp + ct * 512, lane), acc[ct]);
  }

  const int nlo = lane & 15, mhi = (lane >> 4) * 8;
#pragma unroll
  for (int ct = 0; ct < 4; ++ct) {
#pragma unroll
    for (int i = 0; i < 8; ++i) {
      const int t = t0 + mhi + i;
      const int dh = ct * 16 + nlo;
      Ostage[((size_t)b * T_ + t) * D_ + h * DH_ + dh] = (_Float16)acc[ct][i];
    }
  }
}

// ---------------------------------------------------------------------------
// Kernel E: out = Ostage @ Wo (fp32 to d_out O section). A already f16
// row-major (direct fragment loads); Wo f32 staged/transposed through LDS.
// Final fp32 output is write-only -> non-temporal stores.
// ---------------------------------------------------------------------------
__global__ __launch_bounds__(256) void k_out(
    const _Float16* __restrict__ Ostage, const float* __restrict__ Wo,
    float* __restrict__ out) {
  const int n0 = blockIdx.x * 128;
  const int m0 = blockIdx.y * 128;
  const int tid = threadIdx.x, lane = tid & 31, wid = tid >> 5;
  const int wr = wid >> 1, wc = wid & 1;

  __shared__ __align__(16) _Float16 Bs[128 * 40];

  v8f acc[2][4] = {};

  for (int k0 = 0; k0 < D_; k0 += 32) {
#pragma unroll
    for (int c = 0; c < 4; ++c) {
      int e = c * 1024 + tid * 4;
      int kr = e >> 7, col = e & 127;
      const float4 f = *(const float4*)(Wo + (size_t)(k0 + kr) * D_ + n0 + col);
      Bs[(col + 0) * 40 + kr] = (_Float16)f.x;
      Bs[(col + 1) * 40 + kr] = (_Float16)f.y;
      Bs[(col + 2) * 40 + kr] = (_Float16)f.z;
      Bs[(col + 3) * 40 + kr] = (_Float16)f.w;
    }
    __syncthreads();
    v16h af0 = load_a_rm(Ostage + (size_t)(m0 + wr * 32) * D_, D_, k0, lane);
    v16h af1 = load_a_rm(Ostage + (size_t)(m0 + wr * 32 + 16) * D_, D_, k0, lane);
#pragma unroll
    for (int ct = 0; ct < 4; ++ct) {
      v16h bf = load_b_t(Bs, 40, wc * 64 + ct * 16, lane);
      acc[0][ct] = wmma32(af0, bf, acc[0][ct]);
      acc[1][ct] = wmma32(af1, bf, acc[1][ct]);
    }
    __syncthreads();
  }

  const int nlo = lane & 15, mhi = (lane >> 4) * 8;
#pragma unroll
  for (int rt = 0; rt < 2; ++rt) {
#pragma unroll
    for (int ct = 0; ct < 4; ++ct) {
#pragma unroll
      for (int i = 0; i < 8; ++i) {
        const int mrow = m0 + wr * 32 + rt * 16 + mhi + i;
        const int ncol = n0 + wc * 64 + ct * 16 + nlo;
        __builtin_nontemporal_store(acc[rt][ct][i], &out[(size_t)mrow * D_ + ncol]);
      }
    }
  }
}

// ---------------------------------------------------------------------------
extern "C" void kernel_launch(void* const* d_in, const int* in_sizes, int n_in,
                              void* d_out, int out_size, void* d_ws, size_t ws_size,
                              hipStream_t stream) {
  (void)in_sizes; (void)n_in; (void)out_size; (void)ws_size;
  const float* qkv  = (const float*)d_in[0];
  const int*   mask = (const int*)d_in[1];
  const float* pos  = (const float*)d_in[2];
  const float* Wq   = (const float*)d_in[3];
  const float* Wk   = (const float*)d_in[4];
  const float* Wv   = (const float*)d_in[5];
  const float* Wpos = (const float*)d_in[6];
  const float* bpos = (const float*)d_in[7];
  const float* Wo   = (const float*)d_in[8];
  const float* u    = (const float*)d_in[9];
  const float* v    = (const float*)d_in[10];

  float* out = (float*)d_out;                      // O: (B,T,D) fp32
  float* att_out = out + (size_t)B_ * T_ * D_;     // att_w: (B,H,T,T) fp32

  // Workspace carve (f16): ~172 MB total
  char* w = (char*)d_ws;
  const size_t QU = (size_t)B_ * H_ * T_ * DH_;          // 4,194,304 halves
  const size_t ETBH = (size_t)H_ * LP16_ * 2 * 32 * 16;  // 2,129,920 halves
  const size_t ATT = (size_t)B_ * H_ * T_ * T_;          // 67,108,864 halves
  _Float16* Qu     = (_Float16*)w;                 w += QU * 2;
  _Float16* Qv     = (_Float16*)w;                 w += QU * 2;
  _Float16* KB     = (_Float16*)w;                 w += QU * 2;   // same element count
  _Float16* VB     = (_Float16*)w;                 w += QU * 2;
  _Float16* ETB    = (_Float16*)w;                 w += ETBH * 2;
  _Float16* attH   = (_Float16*)w;                 w += ATT * 2;
  _Float16* Ostage = (_Float16*)w;                 /* + QU*2 */

  // Zero the padded E buffer so the skewed Rel strip reads zeros out of range.
  hipMemsetAsync(ETB, 0, ETBH * 2, stream);

  k_proj_qkv<<<dim3(D_ / 128, (B_ * T_) / 128, 3), 256, 0, stream>>>(
      qkv, Wq, Wk, Wv, u, v, Qu, Qv, KB, VB);

  k_proj_e<<<dim3(D_ / 128, 2048 / 128, 1), 256, 0, stream>>>(pos, Wpos, bpos, ETB);

  const int smemC = (16 * RELW_ + 16 * T_) * (int)sizeof(float);  // 133,120 B
  hipFuncSetAttribute((const void*)k_scores_softmax,
                      hipFuncAttributeMaxDynamicSharedMemorySize, smemC);
  k_scores_softmax<<<dim3(T_ / 16, H_, B_), 256, smemC, stream>>>(
      Qu, Qv, KB, ETB, mask, att_out, attH);

  k_attv<<<dim3(T_ / 128, H_, B_), 256, 0, stream>>>(attH, VB, Ostage);

  k_out<<<dim3(D_ / 128, (B_ * T_) / 128, 1), 256, 0, stream>>>(Ostage, Wo, out);
}
